// MultiHeadAttention_83923660964113
// MI455X (gfx1250) — compile-verified, matrix operand order checked by reading, not run on hardware
//
#include <hip/hip_runtime.h>

// ---------------------------------------------------------------------------
// CDNA5 (gfx1250) SRA-attention, bf16 WMMA pipeline, wave32.
// ---------------------------------------------------------------------------

typedef __attribute__((ext_vector_type(16))) __bf16        bf16x16;
typedef __attribute__((ext_vector_type(8)))  __bf16        bf16x8;
typedef __attribute__((ext_vector_type(8)))  float         f32x8;
typedef __attribute__((ext_vector_type(4)))  unsigned int  u32x4;

#define B_  2
#define N_  5000
#define C_  256
#define H_  8
#define M_  2500
#define DK_ 32
#define XT_STRIDE 40   // bf16 elems per LDS row: 16B-aligned, 20-dword pitch

__device__ __forceinline__ f32x8 fzero8() {
    f32x8 z;
#pragma unroll
    for (int i = 0; i < 8; ++i) z[i] = 0.0f;
    return z;
}
__device__ __forceinline__ bf16x8 bzero8() {
    bf16x8 z;
#pragma unroll
    for (int i = 0; i < 8; ++i) z[i] = (__bf16)0.0f;
    return z;
}

__device__ __forceinline__ f32x8 wmma_bf16(bf16x16 a, bf16x16 b, f32x8 c) {
    return __builtin_amdgcn_wmma_f32_16x16x32_bf16(
        false, a, false, b, (short)0, c, false, false);
}

// K-contiguous fragment: element i -> k = (i<8 ? i : i+8) + 8*khalf.
// Two 16-byte chunks at +8*khalf and +16+8*khalf.
__device__ __forceinline__ bf16x16 load_frag_kc(const __bf16* p, int khalf) {
    const __bf16* q = p + khalf * 8;
    bf16x8 lo = *(const bf16x8*)(q);
    bf16x8 hi = *(const bf16x8*)(q + 16);
    bf16x16 r;
#pragma unroll
    for (int i = 0; i < 8; ++i) { r[i] = lo[i]; r[i + 8] = hi[i]; }
    return r;
}

// Scatter two staged bf16x8 row-vectors (rows nl, nl+1 of a [c][n] LDS tile)
// as packed b32 stores: one V_PERM_B32 per dword instead of and/or/mov chains.
__device__ __forceinline__ void scatter_pair(__bf16* base /* &tile[(cc*8)*STRIDE + 2*nlh] */,
                                             bf16x8 v_lo, bf16x8 v_hi) {
    u32x4 lo = __builtin_bit_cast(u32x4, v_lo);
    u32x4 hi = __builtin_bit_cast(u32x4, v_hi);
#pragma unroll
    for (int d = 0; d < 4; ++d) {
        unsigned int pk0 = __builtin_amdgcn_perm(hi[d], lo[d], 0x05040100u);
        unsigned int pk1 = __builtin_amdgcn_perm(hi[d], lo[d], 0x07060302u);
        *(unsigned int*)(base + (2 * d)     * XT_STRIDE) = pk0;
        *(unsigned int*)(base + (2 * d + 1) * XT_STRIDE) = pk1;
    }
}

// ---------------------------------------------------------------------------
// Kernel 0: f32 -> bf16 conversion, zero-padding [n_in, n_out)
// ---------------------------------------------------------------------------
__global__ void k_cvt_bf16(const float* __restrict__ in, __bf16* __restrict__ out,
                           int n_in, int n_out) {
    int i = blockIdx.x * blockDim.x + threadIdx.x;
    int stride = gridDim.x * blockDim.x;
    for (; i < n_out; i += stride)
        out[i] = (i < n_in) ? (__bf16)in[i] : (__bf16)0.0f;
}

// ---------------------------------------------------------------------------
// Kernel 1: xr[b,m,:] = LayerNorm( Wc @ x + bc )   (bf16 out)
// 64 m-rows x 256 cols per WG; double-buffered, software-pipelined x^T tile.
// ---------------------------------------------------------------------------
__global__ void __launch_bounds__(256)
k_reduce_ln(const __bf16* __restrict__ xbf,   // (B,N,C)
            const __bf16* __restrict__ wcbf,  // (M,N) + zero pad
            const float*  __restrict__ bc,
            const float*  __restrict__ gamma,
            const float*  __restrict__ beta,
            __bf16*       __restrict__ xr)    // (B,M,C)
{
    __shared__ __attribute__((aligned(16))) __bf16 xt[2][C_ * XT_STRIDE]; // [c][n] tiles
    __shared__ float partials[4][16][2][2];

    const int b   = blockIdx.y;
    const int m0  = blockIdx.x * 64;
    const int tid = threadIdx.x;
    const int w = tid >> 5, lane = tid & 31;
    const int rb = w & 3, ch = w >> 2;
    const int col16 = lane & 15, khalf = lane >> 4, rowbase = khalf * 8;

    // staging mapping: lane<16 -> even cc, lane>=16 -> odd cc (conflict-free b32 scatter)
    const int s_half = lane >> 4;
    const int s_nlh  = lane & 15;             // n-row pair index (rows 2*s_nlh, +1)
    const __bf16* xbase = xbf + (size_t)b * N_ * C_;

    bf16x8 st[4];                             // staged x rows (2 passes x row pair)
    auto stage_load = [&](int n0) {
        const int row = 2 * s_nlh;
#pragma unroll
        for (int q = 0; q < 2; ++q) {
            const int cc = (w + q * 8) * 2 + s_half;      // 0..31
            const __bf16* p = xbase + (size_t)(n0 + row) * C_ + cc * 8;
            st[q * 2 + 0] = ((n0 + row)     < N_) ? *(const bf16x8*)(p)      : bzero8();
            st[q * 2 + 1] = ((n0 + row + 1) < N_) ? *(const bf16x8*)(p + C_) : bzero8();
        }
    };
    auto stage_store = [&](int buf) {
#pragma unroll
        for (int q = 0; q < 2; ++q) {
            const int cc = (w + q * 8) * 2 + s_half;
            scatter_pair(&xt[buf][(cc * 8) * XT_STRIDE + 2 * s_nlh],
                         st[q * 2 + 0], st[q * 2 + 1]);
        }
    };

    f32x8 acc[8];
#pragma unroll
    for (int t = 0; t < 8; ++t) acc[t] = fzero8();

    const int mrow = min(m0 + rb * 16 + col16, M_ - 1);
    const __bf16* wc_row = wcbf + (size_t)mrow * N_;

    const int NSTEPS = (N_ + 31) / 32;        // 157
    stage_load(0);
    stage_store(0);
    __syncthreads();

    for (int it = 0; it < NSTEPS; ++it) {
        const int n0  = it * 32;
        const int cur = it & 1;
        const bool have_next = (it + 1) < NSTEPS;
        if (have_next) {
            stage_load(n0 + 32);
            __builtin_prefetch(wc_row + n0 + 64, 0, 1);
        }

        bf16x16 afrag = load_frag_kc(wc_row + n0, khalf);
        bf16x16 bfr[8];
#pragma unroll
        for (int t = 0; t < 8; ++t)
            bfr[t] = load_frag_kc(&xt[cur][(ch * 128 + t * 16 + col16) * XT_STRIDE], khalf);
#pragma unroll
        for (int t = 0; t < 8; ++t)
            acc[t] = wmma_bf16(afrag, bfr[t], acc[t]);

        if (have_next) stage_store(cur ^ 1);
        __syncthreads();
    }

    // ---- fused bias + LayerNorm ----
    float bcv[8], s1[8], s2[8];
#pragma unroll
    for (int r = 0; r < 8; ++r) {
        bcv[r] = bc[min(m0 + rb * 16 + rowbase + r, M_ - 1)];
        s1[r] = 0.0f; s2[r] = 0.0f;
    }
#pragma unroll
    for (int t = 0; t < 8; ++t)
#pragma unroll
        for (int r = 0; r < 8; ++r) {
            float v = acc[t][r] + bcv[r];
            acc[t][r] = v;
            s1[r] += v; s2[r] += v * v;
        }
#pragma unroll
    for (int msk = 1; msk < 16; msk <<= 1)
#pragma unroll
        for (int r = 0; r < 8; ++r) {
            s1[r] += __shfl_xor(s1[r], msk, 32);
            s2[r] += __shfl_xor(s2[r], msk, 32);
        }
    if (col16 == 0) {
#pragma unroll
        for (int r = 0; r < 8; ++r) {
            partials[rb][rowbase + r][ch][0] = s1[r];
            partials[rb][rowbase + r][ch][1] = s2[r];
        }
    }
    __syncthreads();

    float mu[8], rstd[8];
#pragma unroll
    for (int r = 0; r < 8; ++r) {
        float su = partials[rb][rowbase + r][0][0] + partials[rb][rowbase + r][1][0];
        float sq = partials[rb][rowbase + r][0][1] + partials[rb][rowbase + r][1][1];
        float mean = su * (1.0f / 256.0f);
        float var  = sq * (1.0f / 256.0f) - mean * mean;
        mu[r] = mean;
        rstd[r] = rsqrtf(var + 1e-5f);
    }
#pragma unroll
    for (int t = 0; t < 8; ++t) {
        const int c = ch * 128 + t * 16 + col16;
        const float g = gamma[c], be = beta[c];
#pragma unroll
        for (int r = 0; r < 8; ++r) {
            const int mm = m0 + rb * 16 + rowbase + r;
            if (mm < M_) {
                float v = g * (acc[t][r] - mu[r]) * rstd[r] + be;
                xr[((size_t)b * M_ + mm) * C_ + c] = (__bf16)v;
            }
        }
    }
}

// ---------------------------------------------------------------------------
// Kernel 2: Out[r,c] = sum_k A[r,k]*W[c,k] + bias[c]  (Q/K/V projections)
// Both operands K-contiguous -> register-fed WMMA, no LDS.
// ---------------------------------------------------------------------------
__global__ void __launch_bounds__(256)
k_proj(const __bf16* __restrict__ A,
       const __bf16* __restrict__ Wt,
       const float*  __restrict__ bias,
       __bf16*       __restrict__ Out,
       int rows)
{
    const int tid = threadIdx.x;
    const int w = tid >> 5, lane = tid & 31;
    const int col16 = lane & 15, khalf = lane >> 4, rowbase = khalf * 8;
    const int r0 = blockIdx.x * 128 + w * 16;
    const int c0 = blockIdx.y * 64;

    const int arow = min(r0 + col16, rows - 1);
    const __bf16* ap = A + (size_t)arow * C_;

    f32x8 acc[4];
#pragma unroll
    for (int t = 0; t < 4; ++t) acc[t] = fzero8();

#pragma unroll
    for (int k0 = 0; k0 < C_; k0 += 32) {
        bf16x16 afrag = load_frag_kc(ap + k0, khalf);
        bf16x16 bfr[4];
#pragma unroll
        for (int t = 0; t < 4; ++t)
            bfr[t] = load_frag_kc(Wt + (size_t)(c0 + t * 16 + col16) * C_ + k0, khalf);
#pragma unroll
        for (int t = 0; t < 4; ++t)
            acc[t] = wmma_bf16(afrag, bfr[t], acc[t]);
    }
#pragma unroll
    for (int t = 0; t < 4; ++t) {
        const int c = c0 + t * 16 + col16;
        const float bv = bias[c];
#pragma unroll
        for (int r = 0; r < 8; ++r) {
            const int row = r0 + rowbase + r;
            if (row < rows) Out[(size_t)row * C_ + c] = (__bf16)(acc[t][r] + bv);
        }
    }
}

// ---------------------------------------------------------------------------
// Kernel 3: flash attention. grid = (N/128, H, B); 8 waves x 16 q-rows.
// Double-buffered V^T tile; one barrier per 32-m step.
// ---------------------------------------------------------------------------
__global__ void __launch_bounds__(256)
k_attn(const __bf16* __restrict__ Q,
       const __bf16* __restrict__ K,
       const __bf16* __restrict__ V,
       float*        __restrict__ Out)
{
    __shared__ __attribute__((aligned(16))) __bf16 vt[2][32 * XT_STRIDE];  // [d][m]
    __shared__ __attribute__((aligned(16))) __bf16 pbuf[8][16 * XT_STRIDE];

    const int tid = threadIdx.x;
    const int w = tid >> 5, lane = tid & 31;
    const int col16 = lane & 15, khalf = lane >> 4, rowbase = khalf * 8;
    const int b = blockIdx.z, h = blockIdx.y;
    const int n0 = blockIdx.x * 128 + w * 16;

    const __bf16* Kbase = K + (size_t)b * M_ * C_ + h * DK_;
    const __bf16* Vbase = V + (size_t)b * M_ * C_ + h * DK_;

    const int qrow = min(n0 + col16, N_ - 1);
    const bf16x16 qfrag = load_frag_kc(Q + ((size_t)b * N_ + qrow) * C_ + h * DK_, khalf);

    // V staging: threads 0..63; mlh = m-row pair, cc = d-chunk (conflict-free b32 scatter)
    const int v_mlh = tid & 15;
    const int v_cc  = (tid >> 4) & 3;
    bf16x8 stv0, stv1;
    auto vload = [&](int mb) {
        if (tid < 64) {
            const int m = mb + 2 * v_mlh;
            const __bf16* p = Vbase + (size_t)m * C_ + v_cc * 8;
            stv0 = (m     < M_) ? *(const bf16x8*)(p)      : bzero8();
            stv1 = (m + 1 < M_) ? *(const bf16x8*)(p + C_) : bzero8();
        }
    };
    auto vstore = [&](int buf) {
        if (tid < 64)
            scatter_pair(&vt[buf][(v_cc * 8) * XT_STRIDE + 2 * v_mlh], stv0, stv1);
    };

    f32x8 o0 = fzero8(), o1 = fzero8();
    float mrow[8], lrow[8];
#pragma unroll
    for (int r = 0; r < 8; ++r) { mrow[r] = -1e30f; lrow[r] = 0.0f; }

    const float scale = 0.17677669529663687f; // 1/sqrt(32)
    const int MSTEPS = (M_ + 31) / 32;        // 79

    vload(0);
    vstore(0);
    __syncthreads();

    for (int it = 0; it < MSTEPS; ++it) {
        const int m0  = it * 32;
        const int cur = it & 1;
        const bool have_next = (it + 1) < MSTEPS;
        if (have_next) vload(m0 + 32);

        // ---- scores: two 16x16 tiles, one K=32 WMMA each ----
        const int mc0 = min(m0 + col16,      M_ - 1);
        const int mc1 = min(m0 + 16 + col16, M_ - 1);
        bf16x16 kf0 = load_frag_kc(Kbase + (size_t)mc0 * C_, khalf);
        bf16x16 kf1 = load_frag_kc(Kbase + (size_t)mc1 * C_, khalf);
        f32x8 s[2];
        s[0] = wmma_bf16(qfrag, kf0, fzero8());
        s[1] = wmma_bf16(qfrag, kf1, fzero8());
        const bool valid0 = (m0 + col16)      < M_;
        const bool valid1 = (m0 + 16 + col16) < M_;
#pragma unroll
        for (int r = 0; r < 8; ++r) {
            s[0][r] = valid0 ? s[0][r] * scale : -1e30f;
            s[1][r] = valid1 ? s[1][r] * scale : -1e30f;
        }

        // ---- online softmax (row stats replicated across 16-lane halves) ----
        float mx[8];
#pragma unroll
        for (int r = 0; r < 8; ++r) mx[r] = fmaxf(s[0][r], s[1][r]);
#pragma unroll
        for (int msk = 1; msk < 16; msk <<= 1)
#pragma unroll
            for (int r = 0; r < 8; ++r) mx[r] = fmaxf(mx[r], __shfl_xor(mx[r], msk, 32));

        float alpha[8];
#pragma unroll
        for (int r = 0; r < 8; ++r) {
            float mnew = fmaxf(mrow[r], mx[r]);
            alpha[r] = __expf(mrow[r] - mnew);
            mrow[r] = mnew;
        }
        float rs[8];
#pragma unroll
        for (int r = 0; r < 8; ++r) rs[r] = 0.0f;
#pragma unroll
        for (int t = 0; t < 2; ++t)
#pragma unroll
            for (int r = 0; r < 8; ++r) {
                float p = __expf(s[t][r] - mrow[r]);
                s[t][r] = p;
                rs[r] += p;
            }
#pragma unroll
        for (int msk = 1; msk < 16; msk <<= 1)
#pragma unroll
            for (int r = 0; r < 8; ++r) rs[r] += __shfl_xor(rs[r], msk, 32);
#pragma unroll
        for (int r = 0; r < 8; ++r) {
            lrow[r] = lrow[r] * alpha[r] + rs[r];
            o0[r] *= alpha[r];
            o1[r] *= alpha[r];
        }

        // ---- transpose P via per-wave LDS bounce (in-order within wave) ----
        __bf16* pw = pbuf[w];
#pragma unroll
        for (int t = 0; t < 2; ++t)
#pragma unroll
            for (int r = 0; r < 8; ++r)
                pw[(rowbase + r) * XT_STRIDE + t * 16 + col16] = (__bf16)s[t][r];
        __builtin_amdgcn_wave_barrier();

        bf16x16 pfrag = load_frag_kc(&pw[(size_t)col16 * XT_STRIDE], khalf);
        bf16x16 v0 = load_frag_kc(&vt[cur][(size_t)col16 * XT_STRIDE], khalf);
        bf16x16 v1 = load_frag_kc(&vt[cur][(size_t)(16 + col16) * XT_STRIDE], khalf);
        o0 = wmma_bf16(pfrag, v0, o0);
        o1 = wmma_bf16(pfrag, v1, o1);

        if (have_next) vstore(cur ^ 1);
        __syncthreads();
    }

    // ---- normalize + store (B,N,C) f32, coalesced 64B per half-wave ----
#pragma unroll
    for (int r = 0; r < 8; ++r) {
        const int row = n0 + rowbase + r;
        if (row < N_) {
            const float inv = 1.0f / lrow[r];
            float* op = Out + ((size_t)b * N_ + row) * C_ + h * DK_;
            op[col16]      = o0[r] * inv;
            op[16 + col16] = o1[r] * inv;
        }
    }
}

// ---------------------------------------------------------------------------
// Host-side launcher
// ---------------------------------------------------------------------------
extern "C" void kernel_launch(void* const* d_in, const int* in_sizes, int n_in,
                              void* d_out, int out_size, void* d_ws, size_t ws_size,
                              hipStream_t stream) {
    const float* x     = (const float*)d_in[0];
    const float* Wq    = (const float*)d_in[1];
    const float* bq    = (const float*)d_in[2];
    const float* Wk    = (const float*)d_in[3];
    const float* bk    = (const float*)d_in[4];
    const float* Wv    = (const float*)d_in[5];
    const float* bv    = (const float*)d_in[6];
    const float* Wc    = (const float*)d_in[7];
    const float* bc    = (const float*)d_in[8];
    const float* gamma = (const float*)d_in[9];
    const float* beta  = (const float*)d_in[10];

    char* ws = (char*)d_ws;
    size_t off = 0;
    auto alloc = [&](size_t bytes) -> char* {
        char* p = ws + off;
        off += (bytes + 255) & ~(size_t)255;
        return p;
    };
    __bf16* xbf  = (__bf16*)alloc((size_t)B_ * N_ * C_ * 2);
    __bf16* wcbf = (__bf16*)alloc(((size_t)M_ * N_ + 64) * 2);
    __bf16* wqbf = (__bf16*)alloc((size_t)C_ * C_ * 2);
    __bf16* wkbf = (__bf16*)alloc((size_t)C_ * C_ * 2);
    __bf16* wvbf = (__bf16*)alloc((size_t)C_ * C_ * 2);
    __bf16* xrbf = (__bf16*)alloc((size_t)B_ * M_ * C_ * 2);
    __bf16* qbf  = (__bf16*)alloc((size_t)B_ * N_ * C_ * 2);
    __bf16* kbf  = (__bf16*)alloc((size_t)B_ * M_ * C_ * 2);
    __bf16* vbf  = (__bf16*)alloc((size_t)B_ * M_ * C_ * 2);

    auto cvt = [&](const float* in, __bf16* out, int n_in_elems, int n_out_elems) {
        int blocks = (n_out_elems + 255) / 256;
        k_cvt_bf16<<<blocks, 256, 0, stream>>>(in, out, n_in_elems, n_out_elems);
    };
    cvt(x,  xbf,  B_ * N_ * C_, B_ * N_ * C_);
    cvt(Wc, wcbf, M_ * N_,      M_ * N_ + 64);
    cvt(Wq, wqbf, C_ * C_,      C_ * C_);
    cvt(Wk, wkbf, C_ * C_,      C_ * C_);
    cvt(Wv, wvbf, C_ * C_,      C_ * C_);

    k_reduce_ln<<<dim3((M_ + 63) / 64, B_), 256, 0, stream>>>(
        xbf, wcbf, bc, gamma, beta, xrbf);

    k_proj<<<dim3((B_ * N_ + 127) / 128, C_ / 64), 256, 0, stream>>>(
        xbf, wqbf, bq, qbf, B_ * N_);
    k_proj<<<dim3((B_ * M_ + 127) / 128, C_ / 64), 256, 0, stream>>>(
        xrbf, wkbf, bk, kbf, B_ * M_);
    k_proj<<<dim3((B_ * M_ + 127) / 128, C_ / 64), 256, 0, stream>>>(
        xrbf, wvbf, bv, vbf, B_ * M_);

    k_attn<<<dim3((N_ + 127) / 128, H_, B_), 256, 0, stream>>>(
        qbf, kbf, vbf, (float*)d_out);
}
